// MultiHeadAttention_89670327206125
// MI455X (gfx1250) — compile-verified, matrix operand order checked by reading, not run on hardware
//
#include <hip/hip_runtime.h>
#include <hip/hip_bf16.h>

typedef __attribute__((ext_vector_type(16))) __bf16 v16bf;
typedef __attribute__((ext_vector_type(8)))  __bf16 v8bf;
typedef __attribute__((ext_vector_type(8)))  float  v8f;

#define WMMA_BF16(a, b, c) \
    __builtin_amdgcn_wmma_f32_16x16x32_bf16(false, (a), false, (b), (short)0, (c), false, false)

// Problem constants from the reference
constexpr int BATCH = 2;
constexpr int SEQ   = 2048;
constexpr int DIM   = 1024;
constexpr int HEADS_C = 16;
constexpr int DK    = 64;          // DIM / HEADS
constexpr int ROWS  = BATCH * SEQ; // 4096
constexpr float LOG2E  = 1.44269504088896340736f;
constexpr float SCALE2 = 0.125f * LOG2E;  // (1/sqrt(64)) * log2(e): scores scaled into log2 domain
constexpr float MASKVAL2 = -1e9f * LOG2E; // masked score in log2 domain
constexpr float EPS = 1e-5f;

// ---------------------------------------------------------------------------
// helpers
// ---------------------------------------------------------------------------
__device__ inline float fast_exp2(float x) { return __builtin_amdgcn_exp2f(x); }  // v_exp_f32

__device__ inline v16bf load_frag(const __bf16* p0, const __bf16* p1) {
    // elements 0..7 from p0 (K = hl*8 + 0..7), 8..15 from p1 (K = 16 + hl*8 + 0..7)
    v8bf lo = *(const v8bf*)p0;
    v8bf hi = *(const v8bf*)p1;
    v16bf r;
#pragma unroll
    for (int i = 0; i < 8; ++i) { r[i] = lo[i]; r[i + 8] = hi[i]; }
    return r;
}

__device__ inline float rowmax16(float v) {
#pragma unroll
    for (int off = 1; off < 16; off <<= 1) v = fmaxf(v, __shfl_xor(v, off, 16));
    return v;
}
__device__ inline float rowsum16(float v) {
#pragma unroll
    for (int off = 1; off < 16; off <<= 1) v += __shfl_xor(v, off, 16);
    return v;
}

// ---------------------------------------------------------------------------
// conversion kernels
// ---------------------------------------------------------------------------
__global__ void cvt_f32_to_bf16(const float* __restrict__ src, __bf16* __restrict__ dst, int n) {
    for (int i = blockIdx.x * blockDim.x + threadIdx.x; i < n; i += gridDim.x * blockDim.x)
        dst[i] = (__bf16)src[i];
}

// W[k][n] (KxN fp32) -> WT[n][k] (NxK bf16)
__global__ void transpose_w_bf16(const float* __restrict__ W, __bf16* __restrict__ WT, int K, int N) {
    int total = K * N;
    for (int i = blockIdx.x * blockDim.x + threadIdx.x; i < total; i += gridDim.x * blockDim.x) {
        int k = i / N, n = i % N;
        WT[(size_t)n * K + k] = (__bf16)W[i];
    }
}

// ---------------------------------------------------------------------------
// generic WMMA GEMM:  C[M,N] = A[M,K] @ (BT[N,K])^T + bias  (+ resid)
// one wave computes a 16x64 output tile
// ---------------------------------------------------------------------------
__global__ void __launch_bounds__(32)
gemm_wmma(const __bf16* __restrict__ A, const __bf16* __restrict__ BT,
          const float* __restrict__ bias, const float* __restrict__ resid,
          float* __restrict__ outF, __bf16* __restrict__ outB, __bf16* __restrict__ outVT,
          int M, int N, int K) {
    const int lane = threadIdx.x;
    const int l = lane & 15, hl = lane >> 4;
    const int nt64 = blockIdx.x % (N / 64);
    const int mtile = blockIdx.x / (N / 64);
    const int row0 = mtile * 16;

    v8f acc[4] = {};
    for (int kk = 0; kk < K; kk += 32) {
        const __bf16* ap = A + (size_t)(row0 + l) * K + kk + hl * 8;
        v16bf a = load_frag(ap, ap + 16);
#pragma unroll
        for (int nn = 0; nn < 4; ++nn) {
            int col = nt64 * 64 + nn * 16 + l;
            const __bf16* bp = BT + (size_t)col * K + kk + hl * 8;
            v16bf b = load_frag(bp, bp + 16);
            acc[nn] = WMMA_BF16(a, b, acc[nn]);
        }
    }
#pragma unroll
    for (int nn = 0; nn < 4; ++nn) {
#pragma unroll
        for (int j = 0; j < 8; ++j) {
            int row = row0 + j + 8 * hl;
            int col = nt64 * 64 + nn * 16 + l;
            float v = acc[nn][j] + bias[col];
            if (resid) v += resid[(size_t)row * N + col];
            if (outF)  outF[(size_t)row * N + col] = v;
            if (outB)  outB[(size_t)row * N + col] = (__bf16)v;
            if (outVT) {  // head-transposed layout [B, H, DK, SEQ]
                int b_ = row / SEQ, s_ = row % SEQ;
                int h_ = col / DK, d_ = col % DK;
                outVT[(((size_t)(b_ * HEADS_C + h_)) * DK + d_) * SEQ + s_] = (__bf16)v;
            }
        }
    }
}

// ---------------------------------------------------------------------------
// attention: one wave handles 16 query rows of one (batch, head)
// pass 1: online row max/sum (log2 domain).  pass 2: recompute scores, write
// attn probabilities once, ctx += P@V via WMMA (P staged through LDS)
// ---------------------------------------------------------------------------
__global__ void __launch_bounds__(32)
attn_wmma(const __bf16* __restrict__ qb, const __bf16* __restrict__ kb,
          const __bf16* __restrict__ vT, const int* __restrict__ mask,
          float* __restrict__ attnOut, __bf16* __restrict__ ctxb) {
    __shared__ __bf16 pls[16 * 32];  // P tile in A-operand staging layout

    const int lane = threadIdx.x;
    const int l = lane & 15, hl = lane >> 4;
    const int st = blockIdx.x, h = blockIdx.y, b = blockIdx.z;
    const int s0 = st * 16;

    // q A-fragments for the 16 rows (dk = 64 -> two K=32 fragments)
    const __bf16* qp = qb + ((size_t)(b * SEQ + s0 + l)) * DIM + h * DK;
    v16bf aq0 = load_frag(qp + hl * 8,      qp + 16 + hl * 8);
    v16bf aq1 = load_frag(qp + 32 + hl * 8, qp + 48 + hl * 8);

    const size_t kbase = (size_t)b * SEQ * DIM + h * DK;

    float mrow[8], srow[8];
#pragma unroll
    for (int j = 0; j < 8; ++j) { mrow[j] = -INFINITY; srow[j] = 0.f; }

    // ---- pass 1: online softmax statistics (base-2 exponentials) ----
    for (int t = 0; t < SEQ / 16; ++t) {
        const __bf16* kp = kb + kbase + (size_t)(t * 16 + l) * DIM;
        v16bf b0 = load_frag(kp + hl * 8,      kp + 16 + hl * 8);
        v16bf b1 = load_frag(kp + 32 + hl * 8, kp + 48 + hl * 8);
        v8f sc = {};
        sc = WMMA_BF16(aq0, b0, sc);
        sc = WMMA_BF16(aq1, b1, sc);
        const bool msk = (mask[b * SEQ + t * 16 + l] == 0);
#pragma unroll
        for (int j = 0; j < 8; ++j) {
            float s_ = msk ? MASKVAL2 : sc[j] * SCALE2;
            float nm = fmaxf(mrow[j], rowmax16(s_));
            float ts = rowsum16(fast_exp2(s_ - nm));
            srow[j] = srow[j] * fast_exp2(mrow[j] - nm) + ts;
            mrow[j] = nm;
        }
    }

    float rinv[8];
#pragma unroll
    for (int j = 0; j < 8; ++j) rinv[j] = __builtin_amdgcn_rcpf(srow[j]);

    // ---- pass 2: recompute, normalize, emit attn, accumulate ctx ----
    v8f ctx[4] = {};
    const size_t attnBase = (((size_t)(b * HEADS_C + h)) * SEQ) * SEQ;
    for (int t = 0; t < SEQ / 16; t += 2) {
#pragma unroll
        for (int tt = 0; tt < 2; ++tt) {
            const int tc = t + tt;
            const __bf16* kp = kb + kbase + (size_t)(tc * 16 + l) * DIM;
            v16bf b0 = load_frag(kp + hl * 8,      kp + 16 + hl * 8);
            v16bf b1 = load_frag(kp + 32 + hl * 8, kp + 48 + hl * 8);
            v8f sc = {};
            sc = WMMA_BF16(aq0, b0, sc);
            sc = WMMA_BF16(aq1, b1, sc);
            const int col = tc * 16 + l;
            const bool msk = (mask[b * SEQ + col] == 0);
#pragma unroll
            for (int j = 0; j < 8; ++j) {
                float s_ = msk ? MASKVAL2 : sc[j] * SCALE2;
                float p = fast_exp2(s_ - mrow[j]) * rinv[j];
                int row = s0 + j + 8 * hl;
                attnOut[attnBase + (size_t)row * SEQ + col] = p;
                pls[(j + 8 * hl) * 32 + tt * 16 + l] = (__bf16)p;
            }
        }
        __syncthreads();
        const __bf16* pb = pls + l * 32;
        v16bf ap = load_frag(pb + hl * 8, pb + 16 + hl * 8);
#pragma unroll
        for (int c = 0; c < 4; ++c) {
            const __bf16* vp =
                vT + (((size_t)(b * HEADS_C + h)) * DK + c * 16 + l) * SEQ + t * 16 + hl * 8;
            v16bf bv = load_frag(vp, vp + 16);
            ctx[c] = WMMA_BF16(ap, bv, ctx[c]);
        }
        __syncthreads();
    }
#pragma unroll
    for (int c = 0; c < 4; ++c)
#pragma unroll
        for (int j = 0; j < 8; ++j) {
            int row = s0 + j + 8 * hl;
            int col = h * DK + c * 16 + l;
            ctxb[((size_t)(b * SEQ + row)) * DIM + col] = (__bf16)ctx[c][j];
        }
}

// ---------------------------------------------------------------------------
// LayerNorm over last dim (D=1024), one block per row
// ---------------------------------------------------------------------------
__global__ void __launch_bounds__(256)
layernorm_rows(const float* __restrict__ resid, const float* __restrict__ gamma,
               const float* __restrict__ beta, float* __restrict__ out) {
    __shared__ float red[256];
    const int row = blockIdx.x, tid = threadIdx.x;
    const float* x = resid + (size_t)row * DIM;
    float s = 0.f, s2 = 0.f;
    for (int c = tid; c < DIM; c += 256) { float v = x[c]; s += v; s2 += v * v; }
    red[tid] = s; __syncthreads();
    for (int k = 128; k > 0; k >>= 1) { if (tid < k) red[tid] += red[tid + k]; __syncthreads(); }
    const float mean = red[0] / DIM;
    __syncthreads();
    red[tid] = s2; __syncthreads();
    for (int k = 128; k > 0; k >>= 1) { if (tid < k) red[tid] += red[tid + k]; __syncthreads(); }
    const float var = red[0] / DIM - mean * mean;
    const float inv = rsqrtf(var + EPS);
    for (int c = tid; c < DIM; c += 256)
        out[(size_t)row * DIM + c] = (x[c] - mean) * inv * gamma[c] + beta[c];
}

// ---------------------------------------------------------------------------
extern "C" void kernel_launch(void* const* d_in, const int* in_sizes, int n_in,
                              void* d_out, int out_size, void* d_ws, size_t ws_size,
                              hipStream_t stream) {
    const float* Q    = (const float*)d_in[0];
    const float* K    = (const float*)d_in[1];
    const float* V    = (const float*)d_in[2];
    const int*  mask  = (const int*)d_in[3];
    const float* Wq   = (const float*)d_in[4];
    const float* bq   = (const float*)d_in[5];
    const float* Wk   = (const float*)d_in[6];
    const float* bk   = (const float*)d_in[7];
    const float* Wv   = (const float*)d_in[8];
    const float* bv   = (const float*)d_in[9];
    const float* Wo   = (const float*)d_in[10];
    const float* bo   = (const float*)d_in[11];
    const float* gamma = (const float*)d_in[12];
    const float* beta  = (const float*)d_in[13];

    float* normedOut = (float*)d_out;
    float* attnOut   = (float*)d_out + (size_t)ROWS * DIM;  // after [B,S,D] normed

    // workspace carve-out (256B aligned slabs)
    char* ws = (char*)d_ws;
    size_t off = 0;
    auto carve = [&](size_t bytes) -> void* {
        void* p = ws + off;
        off += (bytes + 255) & ~(size_t)255;
        return p;
    };
    const size_t nBSD = (size_t)ROWS * DIM;
    const size_t nW   = (size_t)DIM * DIM;
    __bf16* Qb   = (__bf16*)carve(nBSD * 2);
    __bf16* Kb   = (__bf16*)carve(nBSD * 2);
    __bf16* Vb   = (__bf16*)carve(nBSD * 2);
    __bf16* WqT  = (__bf16*)carve(nW * 2);
    __bf16* WkT  = (__bf16*)carve(nW * 2);
    __bf16* WvT  = (__bf16*)carve(nW * 2);
    __bf16* WoT  = (__bf16*)carve(nW * 2);
    float*  qf32 = (float*)carve(nBSD * 4);
    __bf16* qbh  = (__bf16*)carve(nBSD * 2);
    __bf16* kbh  = (__bf16*)carve(nBSD * 2);
    __bf16* vTb  = (__bf16*)carve(nBSD * 2);
    __bf16* ctxb = (__bf16*)carve(nBSD * 2);
    float*  residF = (float*)carve(nBSD * 4);
    if (off > ws_size) return;  // workspace too small; nothing safe to do

    // 1. precision conversion + weight transposition
    cvt_f32_to_bf16<<<4096, 256, 0, stream>>>(Q, Qb, (int)nBSD);
    cvt_f32_to_bf16<<<4096, 256, 0, stream>>>(K, Kb, (int)nBSD);
    cvt_f32_to_bf16<<<4096, 256, 0, stream>>>(V, Vb, (int)nBSD);
    transpose_w_bf16<<<4096, 256, 0, stream>>>(Wq, WqT, DIM, DIM);
    transpose_w_bf16<<<4096, 256, 0, stream>>>(Wk, WkT, DIM, DIM);
    transpose_w_bf16<<<4096, 256, 0, stream>>>(Wv, WvT, DIM, DIM);
    transpose_w_bf16<<<4096, 256, 0, stream>>>(Wo, WoT, DIM, DIM);

    // 2. projections (one wave per 16x64 tile)
    const int gemmGrid = (ROWS / 16) * (DIM / 64);  // 4096
    gemm_wmma<<<gemmGrid, 32, 0, stream>>>(Qb, WqT, bq, nullptr, qf32, qbh, nullptr, ROWS, DIM, DIM);
    gemm_wmma<<<gemmGrid, 32, 0, stream>>>(Kb, WkT, bk, nullptr, nullptr, kbh, nullptr, ROWS, DIM, DIM);
    gemm_wmma<<<gemmGrid, 32, 0, stream>>>(Vb, WvT, bv, nullptr, nullptr, nullptr, vTb, ROWS, DIM, DIM);

    // 3. attention (writes attn probs once; ctx accumulated in-register)
    attn_wmma<<<dim3(SEQ / 16, HEADS_C, BATCH), 32, 0, stream>>>(qbh, kbh, vTb, mask, attnOut, ctxb);

    // 4. output projection + projected-q residual
    gemm_wmma<<<gemmGrid, 32, 0, stream>>>(ctxb, WoT, bo, qf32, residF, nullptr, nullptr, ROWS, DIM, DIM);

    // 5. LayerNorm
    layernorm_rows<<<ROWS, 256, 0, stream>>>(residF, gamma, beta, normedOut);
}
